// BasicBlock_58205396795416
// MI455X (gfx1250) — matrix operand order repaired and checked by reading, not compiled, not run on hardware
//
#include <hip/hip_runtime.h>

#define N_ACTIVE 200000
#define C_IN     128
#define C_OUT    128
#define K_OFF    27
#define BN_EPS   1e-4f

#define M_TILE     128
#define LDS_STRIDE 136   // halves per row: 128 + 8 pad (272B) -> 4-bank shift per row

typedef __attribute__((ext_vector_type(16))) _Float16 v16h;
typedef __attribute__((ext_vector_type(8)))  _Float16 v8h;
typedef __attribute__((ext_vector_type(8)))  float    v8f;

// ---- CDNA5 async global->LDS path (gated; falls back to sync copies) ----
#if defined(__has_builtin)
#if __has_builtin(__builtin_amdgcn_global_load_async_to_lds_b128)
#define HAVE_ASYNC_LDS 1
#endif
#endif
#ifndef HAVE_ASYNC_LDS
#define HAVE_ASYNC_LDS 0
#endif

#if HAVE_ASYNC_LDS
// builtin signature (from clang diagnostic): (v4i AS1*, v4i AS3*, imm, imm)
typedef __attribute__((__vector_size__(4 * sizeof(int)))) int v4i_vs;
typedef __attribute__((address_space(1))) v4i_vs* gptr_v4i;
typedef __attribute__((address_space(3))) v4i_vs* lptr_v4i;
// generic->AS1 / generic->AS3 via integer casts (AS3 is the low 32 bits of a
// generic LDS address per the aperture mapping)
#define GASYNC_B128(gp, lp)                                                    \
  __builtin_amdgcn_global_load_async_to_lds_b128(                              \
      (gptr_v4i)(unsigned long long)(gp),                                      \
      (lptr_v4i)(unsigned int)(unsigned long long)(lp), 0, 0)
__device__ __forceinline__ void wait_async0() {
  asm volatile("s_wait_asynccnt 0x0" ::: "memory");
}
#else
__device__ __forceinline__ void wait_async0() {}
#endif

// ---------------- conversion kernels ----------------

__global__ void cvt_feat_f16(const float* __restrict__ in,
                             _Float16* __restrict__ out, int n) {
  int i = blockIdx.x * blockDim.x + threadIdx.x;
  int stride = gridDim.x * blockDim.x;
  for (; i < n; i += stride) out[i] = (_Float16)in[i];
}

// W [K][Cin][Cout] fp32 -> Wt16 [K][Cout][Cin] f16 (transpose so B-fragment
// reads are contiguous along K)
__global__ void cvt_w_f16t(const float* __restrict__ W,
                           _Float16* __restrict__ Wt) {
  int k = blockIdx.x;
  const float* Wk  = W  + (size_t)k * C_IN * C_OUT;
  _Float16*    Wtk = Wt + (size_t)k * C_IN * C_OUT;
  for (int e = threadIdx.x; e < C_IN * C_OUT; e += blockDim.x) {
    int co = e >> 7, ci = e & 127;
    Wtk[co * C_IN + ci] = (_Float16)Wk[ci * C_OUT + co];
  }
}

// ---------------- main sparse-conv GEMM (WMMA, double-buffered async) ----

__global__ __launch_bounds__(256)
void spconv_wmma(const _Float16* __restrict__ feat16,
                 const int* __restrict__ nbr,
                 const _Float16* __restrict__ Wt16,
                 float* __restrict__ out) {
  __shared__ __attribute__((aligned(16))) _Float16 lA[2][M_TILE * LDS_STRIDE];
  __shared__ __attribute__((aligned(16))) _Float16 lB[2][C_OUT * LDS_STRIDE];

  const int tid      = threadIdx.x;
  const int lane     = tid & 31;
  const int wave     = tid >> 5;           // 0..7
  const int tileBase = blockIdx.x * M_TILE;

  // wave tiling: 4 M-groups x 2 N-groups; each wave: 2 M-tiles x 4 N-tiles
  const int mBase = (wave >> 1) * 32;
  const int nBase = (wave & 1) * 64;
  const int m     = lane & 15;
  const int kSel  = lane >> 4;             // 0 or 1

  v8f zero = {};
  v8f acc[2][4];
#pragma unroll
  for (int mt = 0; mt < 2; ++mt)
#pragma unroll
    for (int nt = 0; nt < 4; ++nt) acc[mt][nt] = zero;

  // cooperative load pattern: 2 threads per row, 64 halves (128B) each
  const int gRow  = tid >> 1;              // 0..127
  const int gColH = (tid & 1) * 64;        // 0 or 64 (halves)
  const int grow  = tileBase + gRow;

  // ---- issue loads of offset k into LDS buffer s ----
  auto issue_load = [&](int k, int s) {
    int idx = (grow < N_ACTIVE) ? nbr[k * N_ACTIVE + grow] : -1;
    _Float16* dstA = &lA[s][gRow * LDS_STRIDE + gColH];
    if (idx >= 0) {
      const _Float16* src = feat16 + (size_t)idx * C_IN + gColH;
#if HAVE_ASYNC_LDS
#pragma unroll
      for (int j = 0; j < 8; ++j) GASYNC_B128(src + j * 8, dstA + j * 8);
#else
      const uint4* s4 = (const uint4*)src;
      uint4* d4 = (uint4*)dstA;
#pragma unroll
      for (int j = 0; j < 8; ++j) d4[j] = s4[j];
#endif
    } else {
      uint4 z = {0u, 0u, 0u, 0u};
      uint4* d4 = (uint4*)dstA;
#pragma unroll
      for (int j = 0; j < 8; ++j) d4[j] = z;
    }
    const _Float16* srcB =
        Wt16 + (size_t)k * C_IN * C_OUT + gRow * C_IN + gColH;
    _Float16* dstB = &lB[s][gRow * LDS_STRIDE + gColH];
#if HAVE_ASYNC_LDS
#pragma unroll
    for (int j = 0; j < 8; ++j) GASYNC_B128(srcB + j * 8, dstB + j * 8);
#else
    const uint4* s4b = (const uint4*)srcB;
    uint4* d4b = (uint4*)dstB;
#pragma unroll
    for (int j = 0; j < 8; ++j) d4b[j] = s4b[j];
#endif
  };

  issue_load(0, 0);
  wait_async0();
  __syncthreads();

  for (int k = 0; k < K_OFF; ++k) {
    const int cur = k & 1;
    if (k + 1 < K_OFF) issue_load(k + 1, cur ^ 1);  // DMA next tile during WMMAs

    const _Float16* A = &lA[cur][0];
    const _Float16* B = &lB[cur][0];
#pragma unroll
    for (int kb = 0; kb < 4; ++kb) {
      // A fragments: lane holds K = kb*32 + k0..k0+7 and +16..+23, k0 = kSel*8
      v16h afrag[2];
#pragma unroll
      for (int mt = 0; mt < 2; ++mt) {
        const _Float16* ap =
            A + (mBase + mt * 16 + m) * LDS_STRIDE + kb * 32 + kSel * 8;
        v8h alo = *(const v8h*)(ap);
        v8h ahi = *(const v8h*)(ap + 16);
#pragma unroll
        for (int j = 0; j < 8; ++j) {
          afrag[mt][j] = alo[j];
          afrag[mt][8 + j] = ahi[j];
        }
      }
#pragma unroll
      for (int nt = 0; nt < 4; ++nt) {
        // B fragment: lane = column (Cout), K = kb*32 + k0..k0+7 and +16..+23, k0 = kSel*8
        const _Float16* bp =
            B + (nBase + nt * 16 + m) * LDS_STRIDE + kb * 32 + kSel * 8;
        v8h blo = *(const v8h*)(bp);
        v8h bhi = *(const v8h*)(bp + 16);
        v16h bfrag;
#pragma unroll
        for (int j = 0; j < 8; ++j) {
          bfrag[j] = blo[j];
          bfrag[8 + j] = bhi[j];
        }
        acc[0][nt] = __builtin_amdgcn_wmma_f32_16x16x32_f16(
            false, afrag[0], false, bfrag, (short)0, acc[0][nt], false, false);
        acc[1][nt] = __builtin_amdgcn_wmma_f32_16x16x32_f16(
            false, afrag[1], false, bfrag, (short)0, acc[1][nt], false, false);
      }
    }
    wait_async0();
    __syncthreads();
  }

  // ---- store: C layout -> lanes 0-15: M=v, lanes 16-31: M=v+8; N = lane&15
#pragma unroll
  for (int mt = 0; mt < 2; ++mt) {
#pragma unroll
    for (int nt = 0; nt < 4; ++nt) {
#pragma unroll
      for (int v = 0; v < 8; ++v) {
        int row = tileBase + mBase + mt * 16 + v + kSel * 8;
        if (row < N_ACTIVE)
          out[(size_t)row * C_OUT + nBase + nt * 16 + m] = acc[mt][nt][v];
      }
    }
  }
}

// ---------------- deterministic BN (two-stage) + ReLU ----------------

#define BN_ROWS 512

__global__ void bn_partial(const float* __restrict__ x,
                           float* __restrict__ psum, float* __restrict__ psq) {
  __shared__ float s1[256], s2[256];
  int c    = threadIdx.x & 127;
  int half = threadIdx.x >> 7;
  int r    = blockIdx.x * BN_ROWS + half;
  int rEnd = blockIdx.x * BN_ROWS + BN_ROWS;
  if (rEnd > N_ACTIVE) rEnd = N_ACTIVE;
  float sum = 0.f, sq = 0.f;
  for (; r < rEnd; r += 2) {
    float v = x[(size_t)r * C_OUT + c];
    sum += v; sq += v * v;
  }
  s1[threadIdx.x] = sum; s2[threadIdx.x] = sq;
  __syncthreads();
  if (threadIdx.x < 128) {
    psum[blockIdx.x * 128 + c] = s1[c] + s1[c + 128];
    psq [blockIdx.x * 128 + c] = s2[c] + s2[c + 128];
  }
}

__global__ void bn_finalize(const float* __restrict__ psum,
                            const float* __restrict__ psq, int nblk,
                            const float* __restrict__ gamma,
                            const float* __restrict__ beta,
                            float* __restrict__ scale,
                            float* __restrict__ bias) {
  int c = threadIdx.x;  // 128 threads
  float sum = 0.f, sq = 0.f;
  for (int b = 0; b < nblk; ++b) { sum += psum[b * 128 + c]; sq += psq[b * 128 + c]; }
  float mean = sum / (float)N_ACTIVE;
  float var  = sq  / (float)N_ACTIVE - mean * mean;
  float s    = gamma[c] * rsqrtf(var + BN_EPS);
  scale[c] = s;
  bias[c]  = beta[c] - mean * s;
}

__global__ void bn_apply(float* __restrict__ x,
                         const float* __restrict__ scale,
                         const float* __restrict__ bias) {
  int n = N_ACTIVE * C_OUT;
  int stride = gridDim.x * blockDim.x;
  for (int i = blockIdx.x * blockDim.x + threadIdx.x; i < n; i += stride) {
    int c = i & 127;
    float v = x[i] * scale[c] + bias[c];
    x[i] = v > 0.f ? v : 0.f;
  }
}

// ---------------- launch ----------------

extern "C" void kernel_launch(void* const* d_in, const int* in_sizes, int n_in,
                              void* d_out, int out_size, void* d_ws, size_t ws_size,
                              hipStream_t stream) {
  const float* features = (const float*)d_in[0];
  const int*   nbr      = (const int*)d_in[1];
  const float* W        = (const float*)d_in[2];
  const float* gamma    = (const float*)d_in[3];
  const float* beta     = (const float*)d_in[4];
  float* out = (float*)d_out;

  char* ws = (char*)d_ws;
  size_t off = 0;
  _Float16* feat16 = (_Float16*)(ws + off);
  off += (size_t)N_ACTIVE * C_IN * sizeof(_Float16);
  off = (off + 255) & ~(size_t)255;
  _Float16* wt16 = (_Float16*)(ws + off);
  off += (size_t)K_OFF * C_IN * C_OUT * sizeof(_Float16);
  off = (off + 255) & ~(size_t)255;

  int nblk = (N_ACTIVE + BN_ROWS - 1) / BN_ROWS;  // 391
  float* psum = (float*)(ws + off); off += (size_t)nblk * 128 * sizeof(float);
  float* psq  = (float*)(ws + off); off += (size_t)nblk * 128 * sizeof(float);
  float* scale = (float*)(ws + off); off += 128 * sizeof(float);
  float* bias  = (float*)(ws + off); off += 128 * sizeof(float);

  cvt_feat_f16<<<4096, 256, 0, stream>>>(features, feat16, N_ACTIVE * C_IN);
  cvt_w_f16t<<<K_OFF, 256, 0, stream>>>(W, wt16);

  int mblk = (N_ACTIVE + M_TILE - 1) / M_TILE;  // 1563
  spconv_wmma<<<mblk, 256, 0, stream>>>(feat16, nbr, wt16, out);

  bn_partial<<<nblk, 256, 0, stream>>>(out, psum, psq);
  bn_finalize<<<1, 128, 0, stream>>>(psum, psq, nblk, gamma, beta, scale, bias);
  bn_apply<<<2048, 256, 0, stream>>>(out, scale, bias);
}
